// Self_Attention_12292196401874
// MI455X (gfx1250) — matrix operand order repaired and checked by reading, not compiled, not run on hardware
//
#include <hip/hip_runtime.h>
#include <cstdint>
#include <cstddef>

#define B_ 8
#define N_ 4096
#define C_ 128
#define BN_EPS 1e-3f
#define QSCALE 0.08838834764831845f   // 1/sqrt(128)

typedef __attribute__((ext_vector_type(16))) __bf16          bf16x16;
typedef __attribute__((ext_vector_type(8)))  __bf16          bf16x8;
typedef __attribute__((ext_vector_type(8)))  float           f32x8;

static __device__ __forceinline__ f32x8 wmma_bf16(bf16x16 a, bf16x16 b, f32x8 c) {
    return __builtin_amdgcn_wmma_f32_16x16x32_bf16(
        /*neg_a=*/false, a, /*neg_b=*/false, b,
        /*c_mod=*/(short)0, c, /*reuse_a=*/false, /*reuse_b=*/false);
}

static __device__ __forceinline__ bf16x16 cvt2_bf16x16(f32x8 a0, f32x8 a1) {
    bf16x16 t;
#pragma unroll
    for (int i = 0; i < 8; ++i) { t[i] = (__bf16)a0[i]; t[i + 8] = (__bf16)a1[i]; }
    return t;
}

// ---------------------------------------------------------------------------
// Kernel 0: pre-pack weights into bf16 fragment-native B layout, fold 1/sqrt(C)
// into Wq / bq.  Wfrag[frag = mat*32 + ct*4 + ck][lane][16]:
//   lane<16 : column n = ct*16+lane, K rows ck*32 + 0..15
//   lane>=16: same column,           K rows ck*32 + 16..31
// ---------------------------------------------------------------------------
__global__ __launch_bounds__(32) void prep_kernel(
    const float* __restrict__ wq, const float* __restrict__ bq,
    const float* __restrict__ wk, const float* __restrict__ bk,
    const float* __restrict__ wv, const float* __restrict__ bv,
    __bf16* __restrict__ Wfrag, float* __restrict__ Bprep)
{
    const int lane = threadIdx.x & 31;
    const int lo = lane & 15, hi = lane >> 4;
    const int f   = blockIdx.x;        // 0..95
    const int mat = f >> 5;            // 0=Q 1=K 2=V
    const int rem = f & 31;
    const int ct  = rem >> 2;
    const int ck  = rem & 3;

    const float* Ws[3] = { wq, wk, wv };
    const float* W = Ws[mat];
    const float scale = (mat == 0) ? QSCALE : 1.0f;

    const int n   = ct * 16 + lo;
    const int kkb = ck * 32 + hi * 16;
    bf16x16 t;
#pragma unroll
    for (int e = 0; e < 16; ++e)
        t[e] = (__bf16)(W[(size_t)(kkb + e) * C_ + n] * scale);
    *(bf16x16*)(Wfrag + ((size_t)f * 32 + lane) * 16) = t;

    if (lane < 4) {
        const int g = blockIdx.x * 4 + lane;     // 0..383
        const int m = g >> 7, c = g & 127;
        const float* Bs[3] = { bq, bk, bv };
        Bprep[g] = Bs[m][c] * ((m == 0) ? QSCALE : 1.0f);
    }
}

// ---------------------------------------------------------------------------
// Kernel 1: QKV projection.  grid = B*(N/16) blocks, 128 threads (4 waves).
// Q,K stored row-major bf16 (Q pre-scaled via Wfrag); V transposed [b][c][n].
// ---------------------------------------------------------------------------
__global__ __launch_bounds__(128) void qkv_kernel(
    const float* __restrict__ x,
    const __bf16* __restrict__ Wfrag, const float* __restrict__ Bprep,
    __bf16* __restrict__ Qg, __bf16* __restrict__ Kg, __bf16* __restrict__ Vtg)
{
    const int wave = threadIdx.x >> 5;
    const int lane = threadIdx.x & 31;
    const int lo = lane & 15, hi = lane >> 4;
    const int blk = blockIdx.x;
    const int b  = blk >> 8;
    const int rt = blk & 255;
    const int row = rt * 16 + lo;

    bf16x16 af[4];
#pragma unroll
    for (int ck = 0; ck < 4; ++ck) {
        const float* base = x + (size_t)(b * N_ + row) * C_ + ck * 32 + hi * 16;
        af[ck] = cvt2_bf16x16(*(const f32x8*)base, *(const f32x8*)(base + 8));
    }

#pragma unroll
    for (int i = 0; i < 6; ++i) {
        const int t   = wave * 6 + i;
        const int mat = t >> 3;
        const int ct  = t & 7;
        const int n   = ct * 16 + lo;

        f32x8 acc = {};
#pragma unroll
        for (int ck = 0; ck < 4; ++ck) {
            const int f = mat * 32 + ct * 4 + ck;
            const bf16x16 bt =
                *(const bf16x16*)(Wfrag + ((size_t)f * 32 + lane) * 16);
            acc = wmma_bf16(af[ck], bt, acc);
        }

        const float bias = Bprep[mat * C_ + n];
#pragma unroll
        for (int r = 0; r < 8; ++r) acc[r] += bias;

        if (mat < 2) {
            __bf16* O = (mat == 0) ? Qg : Kg;
#pragma unroll
            for (int r = 0; r < 8; ++r) {
                const int rowOut = rt * 16 + hi * 8 + r;
                O[(size_t)(b * N_ + rowOut) * C_ + n] = (__bf16)acc[r];
            }
        } else {
            bf16x8 pv;
#pragma unroll
            for (int r = 0; r < 8; ++r) pv[r] = (__bf16)acc[r];
            *(bf16x8*)(Vtg + (size_t)(b * C_ + n) * N_ + rt * 16 + hi * 8) = pv;
        }
    }
}

// ---------------------------------------------------------------------------
// Kernel 2: flash attention + residual + inference BatchNorm.
// One wave per 16-query tile; 32-key chunks, ping-pong double-buffered K
// fragments and early-issued V fragments so global loads overlap the softmax.
// ---------------------------------------------------------------------------
struct FlashState {
    f32x8  Oacc[8];
    float  mrow[8], lrow[8];
};

static __device__ __forceinline__ void loadK(
    const __bf16* __restrict__ Kg, int b, int kc, int lo, int hi,
    bf16x16* fa, bf16x16* fb)
{
#pragma unroll
    for (int ck = 0; ck < 4; ++ck) {
        fa[ck] = *(const bf16x16*)(Kg + (size_t)(b * N_ + kc + lo) * C_
                                      + ck * 32 + hi * 16);
        fb[ck] = *(const bf16x16*)(Kg + (size_t)(b * N_ + kc + 16 + lo) * C_
                                      + ck * 32 + hi * 16);
    }
}

static __device__ __forceinline__ void flash_step(
    const __bf16* __restrict__ Kg, const __bf16* __restrict__ Vtg,
    int b, int kc, int lo, int hi,
    const bf16x16 qf[4], __bf16* myp, FlashState& st,
    const bf16x16* kfa, const bf16x16* kfb,      // current chunk K frags
    bf16x16* nka, bf16x16* nkb)                  // preload target for next
{
    // ---- scores from in-register K fragments ----
    f32x8 s0 = {}, s1 = {};
#pragma unroll
    for (int ck = 0; ck < 4; ++ck) {
        s0 = wmma_bf16(qf[ck], kfa[ck], s0);
        s1 = wmma_bf16(qf[ck], kfb[ck], s1);
    }

    // ---- early-issue: V fragments (this chunk) and K fragments (next) ----
    bf16x16 vf[8];
#pragma unroll
    for (int ct = 0; ct < 8; ++ct)
        vf[ct] = *(const bf16x16*)(Vtg + (size_t)(b * C_ + ct * 16 + lo) * N_
                                       + kc + hi * 16);
    const int kn = (kc + 32) & (N_ - 1);
    loadK(Kg, b, kn, lo, hi, nka, nkb);
    // prefetch the K stream two chunks ahead (global_prefetch_b8)
    __builtin_prefetch(Kg + (size_t)(b * N_ + ((kc + 64) & (N_ - 1)) + lo) * C_, 0, 1);

    // ---- online softmax (rows live in fixed VGPR r + lane-half hi) ----
    float alpha[8];
#pragma unroll
    for (int r = 0; r < 8; ++r) {
        float v0 = s0[r], v1 = s1[r];
        float cm = fmaxf(v0, v1);
#pragma unroll
        for (int off = 1; off < 16; off <<= 1)
            cm = fmaxf(cm, __shfl_xor(cm, off, 16));
        const float mnew = fmaxf(st.mrow[r], cm);
        const float a    = __expf(st.mrow[r] - mnew);
        const float p0   = __expf(v0 - mnew);
        const float p1   = __expf(v1 - mnew);
        float rs = p0 + p1;
#pragma unroll
        for (int off = 1; off < 16; off <<= 1)
            rs += __shfl_xor(rs, off, 16);
        st.lrow[r] = st.lrow[r] * a + rs;
        st.mrow[r] = mnew;
        alpha[r] = a;
        const int rowi = r + 8 * hi;              // C layout: M = r + 8*hi
        myp[rowi * 32 + lo]      = (__bf16)p0;
        myp[rowi * 32 + 16 + lo] = (__bf16)p1;
    }

    asm volatile("s_wait_dscnt 0x0" ::: "memory");

    // read back as A fragment (16 rows x 32 keys)
    const __bf16* ap = myp + lo * 32 + hi * 16;
    const bf16x8 p0v = *(const bf16x8*)ap;
    const bf16x8 p1v = *(const bf16x8*)(ap + 8);
    bf16x16 af;
#pragma unroll
    for (int i = 0; i < 8; ++i) { af[i] = p0v[i]; af[i + 8] = p1v[i]; }

    // rescale running output and accumulate P·V
#pragma unroll
    for (int ct = 0; ct < 8; ++ct) {
#pragma unroll
        for (int r = 0; r < 8; ++r) st.Oacc[ct][r] *= alpha[r];
    }
#pragma unroll
    for (int ct = 0; ct < 8; ++ct)
        st.Oacc[ct] = wmma_bf16(af, vf[ct], st.Oacc[ct]);

    asm volatile("s_wait_dscnt 0x0" ::: "memory");   // fence before next scatter
}

__global__ __launch_bounds__(128) void flash_kernel(
    const float* __restrict__ x,
    const __bf16* __restrict__ Qg,
    const __bf16* __restrict__ Kg,
    const __bf16* __restrict__ Vtg,
    const float* __restrict__ gamma, const float* __restrict__ beta,
    const float* __restrict__ mean,  const float* __restrict__ var,
    float* __restrict__ out)
{
    __shared__ __align__(16) __bf16 pbuf[4][16 * 32];

    const int wave = threadIdx.x >> 5;
    const int lane = threadIdx.x & 31;
    const int lo = lane & 15, hi = lane >> 4;
    const int gw = blockIdx.x * 4 + wave;
    const int b  = gw >> 8;
    const int rt = gw & 255;
    const int q0 = rt * 16;

    // Q fragments (A layout), pre-scaled by 1/sqrt(C)
    bf16x16 qf[4];
#pragma unroll
    for (int ck = 0; ck < 4; ++ck) {
        const __bf16* qp = Qg + (size_t)(b * N_ + q0 + lo) * C_ + ck * 32 + hi * 16;
        const bf16x8 a0 = *(const bf16x8*)qp;
        const bf16x8 a1 = *(const bf16x8*)(qp + 8);
        bf16x16 t;
#pragma unroll
        for (int i = 0; i < 8; ++i) { t[i] = a0[i]; t[i + 8] = a1[i]; }
        qf[ck] = t;
    }

    FlashState st;
#pragma unroll
    for (int ct = 0; ct < 8; ++ct) st.Oacc[ct] = f32x8{};
#pragma unroll
    for (int r = 0; r < 8; ++r) { st.mrow[r] = -3.0e38f; st.lrow[r] = 0.0f; }

    __bf16* myp = pbuf[wave];

    // ping-pong K fragment buffers; 64 keys per outer iteration
    bf16x16 ka0[4], kb0[4], ka1[4], kb1[4];
    loadK(Kg, b, 0, lo, hi, ka0, kb0);
    for (int kc = 0; kc < N_; kc += 64) {
        flash_step(Kg, Vtg, b, kc,      lo, hi, qf, myp, st, ka0, kb0, ka1, kb1);
        flash_step(Kg, Vtg, b, kc + 32, lo, hi, qf, myp, st, ka1, kb1, ka0, kb0);
    }

    // ---- epilogue: 1/l, residual, BatchNorm (inference), fp32 store ----
    float rinv[8];
#pragma unroll
    for (int r = 0; r < 8; ++r) rinv[r] = 1.0f / st.lrow[r];

#pragma unroll
    for (int ct = 0; ct < 8; ++ct) {
        const int c = ct * 16 + lo;
        const float iv = gamma[c] * rsqrtf(var[c] + BN_EPS);
        const float sh = beta[c] - mean[c] * iv;
#pragma unroll
        for (int r = 0; r < 8; ++r) {
            const int n = q0 + r + 8 * hi;
            const size_t idx = (size_t)(b * N_ + n) * C_ + c;
            const float o = st.Oacc[ct][r] * rinv[r] + x[idx];
            out[idx] = o * iv + sh;
        }
    }
}

// ---------------------------------------------------------------------------
extern "C" void kernel_launch(void* const* d_in, const int* in_sizes, int n_in,
                              void* d_out, int out_size, void* d_ws, size_t ws_size,
                              hipStream_t stream) {
    const float* x     = (const float*)d_in[0];
    const float* wq    = (const float*)d_in[1];
    const float* bq    = (const float*)d_in[2];
    const float* wk    = (const float*)d_in[3];
    const float* bk    = (const float*)d_in[4];
    const float* wv    = (const float*)d_in[5];
    const float* bv    = (const float*)d_in[6];
    const float* gamma = (const float*)d_in[7];
    const float* beta  = (const float*)d_in[8];
    const float* mmean = (const float*)d_in[9];
    const float* mvar  = (const float*)d_in[10];
    float* out = (float*)d_out;

    const size_t elems = (size_t)B_ * N_ * C_;       // 4.19M
    __bf16* Qg    = (__bf16*)d_ws;
    __bf16* Kg    = Qg + elems;
    __bf16* Vtg   = Kg + elems;
    __bf16* Wfrag = Vtg + elems;                     // 96 frags * 512 bf16 = 96 KB
    float*  Bprep = (float*)(Wfrag + (size_t)96 * 32 * 16);

    prep_kernel<<<dim3(96), dim3(32), 0, stream>>>(
        wq, bq, wk, bk, wv, bv, Wfrag, Bprep);

    qkv_kernel<<<dim3(B_ * (N_ / 16)), dim3(128), 0, stream>>>(
        x, Wfrag, Bprep, Qg, Kg, Vtg);

    flash_kernel<<<dim3(B_ * (N_ / 16) / 4), dim3(128), 0, stream>>>(
        x, Qg, Kg, Vtg, gamma, beta, mmean, mvar, out);
}
